// ModifiedModelB_58007828300485
// MI455X (gfx1250) — matrix operand order repaired and checked by reading, not compiled, not run on hardware
//
#include <hip/hip_runtime.h>
#include <hip/hip_bf16.h>
#include <math.h>

typedef __attribute__((ext_vector_type(16))) _Float16 v16h;
typedef __attribute__((ext_vector_type(8)))  _Float16 v8h;
typedef __attribute__((ext_vector_type(8)))  float    v8f;

__device__ __forceinline__ float relu_f(float x) { return fmaxf(x, 0.f); }

// B-fragment (f16 NHWC: K-elements contiguous) + WMMA for one 16-position tile.
// base = image base + cb + kbB (channel offset). One 32B vector load per frag.
template<int Cin, int H, int W>
__device__ __forceinline__ v8f tile_mma_h(const _Float16* __restrict__ base, int iy, int ix,
                                          v16h af, v8f acc) {
  bool ok = (iy >= 0) && (iy < H) && (ix >= 0) && (ix < W);
  int cy = ok ? iy : 0;
  int cx = ok ? ix : 0;
  v16h bf = *(const v16h*)(base + ((size_t)cy * W + cx) * Cin);
  v16h z = {};
  bf = ok ? bf : z;
  return __builtin_amdgcn_wmma_f32_16x16x32_f16(false, af, false, bf,
                                                (short)0, acc, false, false);
}

// ---------------------------------------------------------------------------
// pack conv weights [Cout,Cin,3,3] f32 -> [Cout][tap][Cin] f16
// ---------------------------------------------------------------------------
__global__ void pack_w_kernel(const float* __restrict__ w, _Float16* __restrict__ wp,
                              int Cout, int Cin) {
  int t = blockIdx.x * blockDim.x + threadIdx.x;
  if (t >= Cout * Cin * 9) return;
  int cin = t % Cin;
  int rest = t / Cin;
  int tap = rest % 9;
  int m = rest / 9;
  wp[t] = (_Float16)w[((size_t)m * Cin + cin) * 9 + tap];
}

// ---------------------------------------------------------------------------
// conv1 (1->32) + relu + 2x2 maxpool, direct VALU; writes f16 NHWC.
// img [16,1,256,256] f32 -> out [16,128,128,32] f16 (flat index == thread id)
// ---------------------------------------------------------------------------
__global__ void conv1_pool_kernel(const float* __restrict__ img,
                                  const float* __restrict__ w,   // [32,1,3,3]
                                  const float* __restrict__ b,
                                  _Float16* __restrict__ out) {
  int t = blockIdx.x * blockDim.x + threadIdx.x;
  int c  = t & 31;
  int px = (t >> 5) & 127;
  int py = (t >> 12) & 127;
  int bi = t >> 19;
  if (bi >= 16) return;
  float wv[9];
#pragma unroll
  for (int k = 0; k < 9; ++k) wv[k] = w[c * 9 + k];
  float bias = b[c];
  const float* im = img + (size_t)bi * 256 * 256;
  float best = 0.f;
#pragma unroll
  for (int sy = 0; sy < 2; ++sy) {
#pragma unroll
    for (int sx = 0; sx < 2; ++sx) {
      int y = 2 * py + sy, x = 2 * px + sx;
      float s = bias;
#pragma unroll
      for (int ky = 0; ky < 3; ++ky) {
        int iy = y + ky - 1;
#pragma unroll
        for (int kx = 0; kx < 3; ++kx) {
          int ix = x + kx - 1;
          float v = (iy >= 0 && iy < 256 && ix >= 0 && ix < 256) ? im[iy * 256 + ix] : 0.f;
          s = fmaf(v, wv[ky * 3 + kx], s);
        }
      }
      best = fmaxf(best, relu_f(s));
    }
  }
  out[t] = (_Float16)best;
}

// ---------------------------------------------------------------------------
// Implicit-GEMM 3x3 SAME conv + relu + fused 2x2 maxpool, f16 NHWC in/out.
// (ky,kx) outer x cin-blocks inner; one A fragment feeds 4 WMMA n-tiles
// (2 conv rows x 32 conv cols -> 16 pooled outputs via LDS).
// ---------------------------------------------------------------------------
template<int Cin, int Cout, int H, int W>
__global__ void conv_pool_wmma_kernel(const _Float16* __restrict__ in,   // [B,H,W,Cin]
                                      const _Float16* __restrict__ wp,   // [Cout,9,Cin]
                                      const float* __restrict__ bias,
                                      _Float16* __restrict__ out) {      // [B,H/2,W/2,Cout]
  constexpr int PH = H >> 1, PW = W >> 1, PW16 = PW >> 4;
  const int lane = threadIdx.x;
  int idx = blockIdx.x;
  const int pxt = idx % PW16; idx /= PW16;
  const int py  = idx % PH;   idx /= PH;
  const int mt  = idx % (Cout >> 4); idx /= (Cout >> 4);
  const int b   = idx;

  const int m0 = mt << 4;
  const int mrow   = lane & 15;
  const int khalfA = (lane >> 4) << 4;   // 0 or 16
  const int ncol   = lane & 15;
  const int kbB    = (lane >> 4) << 4;   // 0 or 16
  const int mh     = lane >> 4;

  const _Float16* apbase = wp + (size_t)(m0 + mrow) * 9 * Cin;
  const _Float16* inb    = in + (size_t)b * H * W * Cin;

  v8f acc0 = {}, acc1 = {}, acc2 = {}, acc3 = {};

  for (int ky = 0; ky < 3; ++ky) {
    for (int kx = 0; kx < 3; ++kx) {
      const int tap = ky * 3 + kx;
      for (int cb = 0; cb < Cin; cb += 32) {
        const _Float16* ap = apbase + tap * Cin + cb + khalfA;
        v8h lo = *(const v8h*)ap;
        v8h hi = *(const v8h*)(ap + 8);
        v16h af = __builtin_shufflevector(lo, hi, 0, 1, 2, 3, 4, 5, 6, 7,
                                          8, 9, 10, 11, 12, 13, 14, 15);
        const int iy0 = 2 * py + ky - 1;
        const int ix0 = (pxt << 5) + ncol + kx - 1;
        const _Float16* basec = inb + cb + kbB;
        acc0 = tile_mma_h<Cin, H, W>(basec, iy0,     ix0,      af, acc0);
        acc1 = tile_mma_h<Cin, H, W>(basec, iy0,     ix0 + 16, af, acc1);
        acc2 = tile_mma_h<Cin, H, W>(basec, iy0 + 1, ix0,      af, acc2);
        acc3 = tile_mma_h<Cin, H, W>(basec, iy0 + 1, ix0 + 16, af, acc3);
      }
    }
  }

  // epilogue: bias + relu -> LDS (2 conv rows x 16 couts x 32 cols) -> 2x2 max
  __shared__ float tile[2][16][32];
#pragma unroll
  for (int r = 0; r < 8; ++r) {
    int m = (mh << 3) + r;
    float bb = bias[m0 + m];
    tile[0][m][ncol]      = relu_f(acc0[r] + bb);
    tile[0][m][16 + ncol] = relu_f(acc1[r] + bb);
    tile[1][m][ncol]      = relu_f(acc2[r] + bb);
    tile[1][m][16 + ncol] = relu_f(acc3[r] + bb);
  }
  __syncthreads();
  const int px = lane & 15;
  v8h ov;
#pragma unroll
  for (int r = 0; r < 8; ++r) {
    int m = (mh << 3) + r;
    float v = fmaxf(fmaxf(tile[0][m][2 * px], tile[0][m][2 * px + 1]),
                    fmaxf(tile[1][m][2 * px], tile[1][m][2 * px + 1]));
    ov[r] = (_Float16)v;
  }
  _Float16* op = out + (((size_t)(b * PH + py)) * PW + (pxt << 4) + px) * Cout
                 + m0 + (mh << 3);
  *(v8h*)op = ov;
}

// ---------------------------------------------------------------------------
// Implicit-GEMM conv + relu (no pool), f16 NHWC. Wave = 16 couts x 64 pos.
// ---------------------------------------------------------------------------
template<int Cin, int Cout, int H, int W>
__global__ void conv_wmma_kernel(const _Float16* __restrict__ in,   // [B,H,W,Cin]
                                 const _Float16* __restrict__ wp,   // [Cout,9,Cin]
                                 const float* __restrict__ bias,
                                 _Float16* __restrict__ out) {      // [B,H,W,Cout]
  constexpr int HW = H * W, NT = HW >> 6;
  const int lane = threadIdx.x;
  int idx = blockIdx.x;
  const int nt = idx % NT; idx /= NT;
  const int mt = idx % (Cout >> 4); idx /= (Cout >> 4);
  const int b  = idx;

  const int m0 = mt << 4;
  const int n0 = nt << 6;
  const int mrow   = lane & 15;
  const int khalfA = (lane >> 4) << 4;
  const int ncol   = lane & 15;
  const int kbB    = (lane >> 4) << 4;
  const int mh     = lane >> 4;

  const _Float16* apbase = wp + (size_t)(m0 + mrow) * 9 * Cin;
  const _Float16* inb    = in + (size_t)b * HW * Cin;

  int ty[4], tx[4];
#pragma unroll
  for (int t = 0; t < 4; ++t) {
    int n = n0 + (t << 4) + ncol;
    ty[t] = n / W;
    tx[t] = n % W;
  }

  v8f acc0 = {}, acc1 = {}, acc2 = {}, acc3 = {};

  for (int ky = 0; ky < 3; ++ky) {
    for (int kx = 0; kx < 3; ++kx) {
      const int tap = ky * 3 + kx;
      for (int cb = 0; cb < Cin; cb += 32) {
        const _Float16* ap = apbase + tap * Cin + cb + khalfA;
        v8h lo = *(const v8h*)ap;
        v8h hi = *(const v8h*)(ap + 8);
        v16h af = __builtin_shufflevector(lo, hi, 0, 1, 2, 3, 4, 5, 6, 7,
                                          8, 9, 10, 11, 12, 13, 14, 15);
        const _Float16* basec = inb + cb + kbB;
        acc0 = tile_mma_h<Cin, H, W>(basec, ty[0] + ky - 1, tx[0] + kx - 1, af, acc0);
        acc1 = tile_mma_h<Cin, H, W>(basec, ty[1] + ky - 1, tx[1] + kx - 1, af, acc1);
        acc2 = tile_mma_h<Cin, H, W>(basec, ty[2] + ky - 1, tx[2] + kx - 1, af, acc2);
        acc3 = tile_mma_h<Cin, H, W>(basec, ty[3] + ky - 1, tx[3] + kx - 1, af, acc3);
      }
    }
  }

  float bb[8];
#pragma unroll
  for (int r = 0; r < 8; ++r) bb[r] = bias[m0 + (mh << 3) + r];
  v8f accs[4] = {acc0, acc1, acc2, acc3};
#pragma unroll
  for (int t = 0; t < 4; ++t) {
    int n = n0 + (t << 4) + ncol;
    v8h ov;
#pragma unroll
    for (int r = 0; r < 8; ++r) ov[r] = (_Float16)relu_f(accs[t][r] + bb[r]);
    *(v8h*)(out + ((size_t)b * HW + n) * Cout + m0 + (mh << 3)) = ov;
  }
}

// ---------------------------------------------------------------------------
// FC via WMMA: out[16,N] = A[16,K] @ Wt[N,K]^T + bias. (tiny; f32 in/out)
// ---------------------------------------------------------------------------
template<int K, int N, int RELU>
__global__ void fc_wmma_kernel(const float* __restrict__ A,
                               const float* __restrict__ Wt,
                               const float* __restrict__ bias,
                               float* __restrict__ out) {
  const int lane = threadIdx.x;
  const int n0 = blockIdx.x << 4;
  const int mrow   = lane & 15;
  const int khalfA = (lane >> 4) << 3;
  const int ncol   = lane & 15;
  const int kbB    = (lane >> 4) << 4;
  const int mh     = lane >> 4;
  const int n = n0 + ncol;
  const float* wrow = Wt + (size_t)((n < N) ? n : 0) * K;

  v8f acc = {};
  for (int kk = 0; kk < K; kk += 32) {
    v16h af, bf;
#pragma unroll
    for (int e = 0; e < 16; ++e) {
      int kl = kbB + e;
      af[e] = (_Float16)A[(size_t)mrow * K + kk + kl];
    }
#pragma unroll
    for (int e = 0; e < 16; ++e) {
      float v = wrow[kk + kbB + e];
      bf[e] = (_Float16)((n < N) ? v : 0.f);
    }
    acc = __builtin_amdgcn_wmma_f32_16x16x32_f16(false, af, false, bf,
                                                 (short)0, acc, false, false);
  }
  if (n < N) {
    float b2 = bias[n];
#pragma unroll
    for (int r = 0; r < 8; ++r) {
      int row = (mh << 3) + r;
      float v = acc[r] + b2;
      if (RELU) v = relu_f(v);
      out[(size_t)row * N + n] = v;
    }
  }
}

// ---------------------------------------------------------------------------
// mean over 1024 positions, NHWC f16 -> f32 feat. <<<16,256>>>, coalesced.
// ---------------------------------------------------------------------------
__global__ void mean_kernel(const _Float16* __restrict__ x, float* __restrict__ feat) {
  int c = threadIdx.x;
  int b = blockIdx.x;
  const _Float16* p = x + (size_t)b * 1024 * 256 + c;
  float s = 0.f;
  for (int i = 0; i < 1024; ++i) s += (float)p[(size_t)i * 256];
  feat[b * 256 + c] = s * (1.f / 1024.f);
}

__global__ void count_kernel(const float* __restrict__ h, const float* __restrict__ w,
                             const float* __restrict__ b, float* __restrict__ outc) {
  int t = threadIdx.x;
  if (t < 16) {
    float s = b[0];
    for (int i = 0; i < 256; ++i) s = fmaf(h[t * 256 + i], w[i], s);
    outc[t] = s;
  }
}

__global__ void norm_kernel(const float* __restrict__ coords, float* __restrict__ cn) {
  int t = blockIdx.x * blockDim.x + threadIdx.x;
  if (t >= 16 * 350) return;
  float x = coords[2 * t], y = coords[2 * t + 1];
  float nrm = fmaxf(sqrtf(x * x + y * y), 1e-12f);
  cn[2 * t] = x / nrm;
  cn[2 * t + 1] = y / nrm;
}

__global__ void mask_kernel(const unsigned char* __restrict__ masks, float* __restrict__ validb) {
  int b = threadIdx.x;
  if (b < 16) {
    int cnt = 0;
    for (int i = 0; i < 350; ++i) cnt += (masks[b * 350 + i] != 0);
    validb[b] = (cnt > 1) ? 1.f : 0.f;
  }
}

// ---------------------------------------------------------------------------
// adjacency: sim -> 1->32->16->1 MLP -> sigmoid -> pair mask. One wave per
// 16x16 (i,j) tile; MLP weights staged in LDS (broadcast reads).
// ---------------------------------------------------------------------------
__global__ void adjacency_kernel(const float* __restrict__ cn,
                                 const unsigned char* __restrict__ masks,
                                 const float* __restrict__ validb,
                                 const float* __restrict__ m1w, const float* __restrict__ m1b,
                                 const float* __restrict__ m2w, const float* __restrict__ m2b,
                                 const float* __restrict__ m3w, const float* __restrict__ m3b,
                                 float* __restrict__ adj) {
  __shared__ float s1w[32], s1b[32], s2w[512], s2b[16], s3w[16];
  int lane = threadIdx.x;
  s1w[lane] = m1w[lane];
  s1b[lane] = m1b[lane];
  for (int i = lane; i < 512; i += 32) s2w[i] = m2w[i];
  if (lane < 16) { s2b[lane] = m2b[lane]; s3w[lane] = m3w[lane]; }
  __syncthreads();
  float b3 = m3b[0];

  int bid = blockIdx.x;
  int jt = bid % 22; bid /= 22;
  int it = bid % 22; bid /= 22;
  int b = bid;
  int i0 = it << 4, j0 = jt << 4;
  float vb = validb[b];

#pragma unroll
  for (int r = 0; r < 8; ++r) {
    int e = lane + (r << 5);
    int i = i0 + (e >> 4);
    int j = j0 + (e & 15);
    if (i < 350 && j < 350) {
      const float* ci = cn + ((size_t)b * 350 + i) * 2;
      const float* cj = cn + ((size_t)b * 350 + j) * 2;
      float sim = fmaf(ci[0], cj[0], ci[1] * cj[1]);
      float h2[16];
#pragma unroll
      for (int q = 0; q < 16; ++q) h2[q] = s2b[q];
      for (int k = 0; k < 32; ++k) {
        float h1 = relu_f(fmaf(sim, s1w[k], s1b[k]));
#pragma unroll
        for (int q = 0; q < 16; ++q) h2[q] = fmaf(h1, s2w[q * 32 + k], h2[q]);
      }
      float a3 = b3;
#pragma unroll
      for (int q = 0; q < 16; ++q) a3 = fmaf(relu_f(h2[q]), s3w[q], a3);
      float corr = 1.f / (1.f + expf(-a3));
      float mv = (masks[b * 350 + i] && masks[b * 350 + j]) ? vb : 0.f;
      adj[((size_t)b * 350 + i) * 350 + j] = corr * mv;
    }
  }
}

// ---------------------------------------------------------------------------
extern "C" void kernel_launch(void* const* d_in, const int* in_sizes, int n_in,
                              void* d_out, int out_size, void* d_ws, size_t ws_size,
                              hipStream_t stream) {
  const float* images = (const float*)d_in[0];
  const unsigned char* node_masks = (const unsigned char*)d_in[1];
  const float* c1w = (const float*)d_in[2];  const float* c1b = (const float*)d_in[3];
  const float* c2w = (const float*)d_in[4];  const float* c2b = (const float*)d_in[5];
  const float* c3w = (const float*)d_in[6];  const float* c3b = (const float*)d_in[7];
  const float* c4w = (const float*)d_in[8];  const float* c4b = (const float*)d_in[9];
  const float* nd1w = (const float*)d_in[10]; const float* nd1b = (const float*)d_in[11];
  const float* nd2w = (const float*)d_in[12]; const float* nd2b = (const float*)d_in[13];
  const float* cp1w = (const float*)d_in[14]; const float* cp1b = (const float*)d_in[15];
  const float* cp2w = (const float*)d_in[16]; const float* cp2b = (const float*)d_in[17];
  const float* mc1w = (const float*)d_in[18]; const float* mc1b = (const float*)d_in[19];
  const float* mc2w = (const float*)d_in[20]; const float* mc2b = (const float*)d_in[21];
  const float* mc3w = (const float*)d_in[22]; const float* mc3b = (const float*)d_in[23];

  // f16 workspace region
  _Float16* hws  = (_Float16*)d_ws;
  _Float16* act0 = hws;                    // [16,128,128,32]  8,388,608 h
  _Float16* act1 = act0 + 8388608;         // [16,64,64,64]    4,194,304 h
  _Float16* act2 = act1 + 4194304;         // [16,32,32,128]   2,097,152 h
  _Float16* act3 = act2 + 2097152;         // [16,32,32,256]   4,194,304 h
  _Float16* wp2  = act3 + 4194304;         // 64*9*32   = 18,432 h
  _Float16* wp3  = wp2 + 18432;            // 128*9*64  = 73,728 h
  _Float16* wp4  = wp3 + 73728;            // 256*9*128 = 294,912 h
  // f32 workspace region (offset is even => 4B aligned)
  float* f32ws  = (float*)(wp4 + 294912);
  float* feat   = f32ws;                   // [16,256]
  float* ndh    = feat + 4096;             // [16,512]
  float* cph    = ndh + 8192;              // [16,256]
  float* cnb    = cph + 4096;              // [16,350,2]
  float* validb = cnb + 11200;             // [16]

  float* coords = (float*)d_out;           // [16,350,2]
  float* adj    = coords + 11200;          // [16,350,350]
  float* cnt    = adj + 1960000;           // [16]

  // weight packing (f32 OIHW -> f16 [Cout][tap][Cin])
  pack_w_kernel<<<(64 * 288 + 255) / 256, 256, 0, stream>>>(c2w, wp2, 64, 32);
  pack_w_kernel<<<(128 * 576 + 255) / 256, 256, 0, stream>>>(c3w, wp3, 128, 64);
  pack_w_kernel<<<(256 * 1152 + 255) / 256, 256, 0, stream>>>(c4w, wp4, 256, 128);

  // CNN encoder (f16 NHWC pipeline)
  conv1_pool_kernel<<<(16 * 32 * 128 * 128) / 256, 256, 0, stream>>>(images, c1w, c1b, act0);
  conv_pool_wmma_kernel<32, 64, 128, 128><<<16 * 4 * 64 * 4, 32, 0, stream>>>(act0, wp2, c2b, act1);
  conv_pool_wmma_kernel<64, 128, 64, 64><<<16 * 8 * 32 * 2, 32, 0, stream>>>(act1, wp3, c3b, act2);
  conv_wmma_kernel<128, 256, 32, 32><<<16 * 16 * 16, 32, 0, stream>>>(act2, wp4, c4b, act3);
  mean_kernel<<<16, 256, 0, stream>>>(act3, feat);

  // heads
  fc_wmma_kernel<256, 512, 1><<<512 / 16, 32, 0, stream>>>(feat, nd1w, nd1b, ndh);
  fc_wmma_kernel<512, 700, 0><<<(700 + 15) / 16, 32, 0, stream>>>(ndh, nd2w, nd2b, coords);
  fc_wmma_kernel<256, 256, 1><<<256 / 16, 32, 0, stream>>>(feat, cp1w, cp1b, cph);
  count_kernel<<<1, 32, 0, stream>>>(cph, cp2w, cp2b, cnt);

  // adjacency
  norm_kernel<<<(5600 + 255) / 256, 256, 0, stream>>>(coords, cnb);
  mask_kernel<<<1, 32, 0, stream>>>(node_masks, validb);
  adjacency_kernel<<<16 * 22 * 22, 32, 0, stream>>>(cnb, node_masks, validb,
      mc1w, mc1b, mc2w, mc2b, mc3w, mc3b, adj);
}
